// RefineModule_15324443312667
// MI455X (gfx1250) — matrix operand order repaired and checked by reading, not compiled, not run on hardware
//
#include <hip/hip_runtime.h>
#include <math.h>

#define NN 20000
#define NE 320000
#define EPS 1e-6f

typedef __attribute__((ext_vector_type(16))) _Float16 v16h;
typedef __attribute__((ext_vector_type(8)))  float    v8f;

__device__ __forceinline__ v8f zero8() {
  v8f z = {0.f,0.f,0.f,0.f,0.f,0.f,0.f,0.f};
  return z;
}

__device__ __forceinline__ v8f wmma16(v16h a, v16h b, v8f c) {
  // D = A(16x32 f16) * B(32x16 f16) + C(16x16 f32)
  return __builtin_amdgcn_wmma_f32_16x16x32_f16(false, a, false, b, (short)0, c, false, false);
}

// B tile: packed per-lane contiguous 16 halves (32B), lane n=tid&15, k=(tid>>4)*16+j
__device__ __forceinline__ v16h load_b_tile(const _Float16* tiles, int tile, int lane) {
  return *(const v16h*)(tiles + (size_t)tile * 512 + lane * 16);
}

// A tile from LDS row-major [16][32] f16 in contiguous k order (k = kh*16 + j), same order as the packed B tiles:
// lane: row = lane&15; kh = lane>>4; halves = K{kh*16..+15}
__device__ __forceinline__ v16h load_a_lds(const _Float16* s, int lane) {
  int r = lane & 15, kh = lane >> 4;
  const _Float16* row = s + r * 32;
  v16h a;
#pragma unroll
  for (int j = 0; j < 8; j++) a[j] = row[kh * 16 + j];
#pragma unroll
  for (int j = 0; j < 8; j++) a[8 + j] = row[kh * 16 + 8 + j];
  return a;
}

__device__ __forceinline__ float eluf(float x) { return x > 0.f ? x : (expf(x) - 1.f); }

// monotone float->uint key for atomic max over signed floats
__device__ __forceinline__ unsigned fkey(float f) {
  unsigned b = __float_as_uint(f);
  return (b & 0x80000000u) ? ~b : (b | 0x80000000u);
}
__device__ __forceinline__ float fkey_dec(unsigned k) {
  unsigned b = (k & 0x80000000u) ? (k & 0x7fffffffu) : ~k;
  return __uint_as_float(b);
}

// NormSE3 for one degree given per-lane 16 channel magnitudes -> mean and rstd over the row's 32 channels
__device__ __forceinline__ void norm_stats(const float* nr, float& mu, float& rstd) {
  float s1 = 0.f, s2 = 0.f;
#pragma unroll
  for (int j = 0; j < 16; j++) { s1 += nr[j]; s2 += nr[j] * nr[j]; }
  s1 += __shfl_xor(s1, 16);
  s2 += __shfl_xor(s2, 16);
  mu = s1 * (1.f / 32.f);
  float var = s2 * (1.f / 32.f) - mu * mu;
  rstd = rsqrtf(var + EPS);
}

// ---------------- weight packing ----------------
__global__ void pack_b_kernel(const float* __restrict__ W, int rows, int cols,
                              int n0, int k0, _Float16* __restrict__ dst) {
  int lane = threadIdx.x;
  int n = n0 + (lane & 15);
  int kb = k0 + (lane >> 4) * 16;
#pragma unroll
  for (int j = 0; j < 16; j++) {
    int k = kb + j;
    float v = (n < rows && k < cols) ? W[n * cols + k] : 0.f;
    dst[lane * 16 + j] = (_Float16)v;
  }
}

__global__ void init_f_kernel(const float* __restrict__ node0, const float* __restrict__ node1,
                              float* __restrict__ F0, float* __restrict__ F1) {
  int i = blockIdx.x * blockDim.x + threadIdx.x;
  if (i < NN * 32) F0[i] = node0[i];
  else if (i < NN * 128) F1[i - NN * 32] = node1[i - NN * 32];
}

__global__ void zero_kernel(unsigned* __restrict__ p, int n) {
  int i = blockIdx.x * blockDim.x + threadIdx.x;
  if (i < n) p[i] = 0u;
}

// ---------------- per-layer node projections (norm + q/k/v WMMAs) ----------------
__global__ void __launch_bounds__(32)
node_project_kernel(const float* __restrict__ F0, const float* __restrict__ F1,
                    const float* __restrict__ g0v, const float* __restrict__ b0v,
                    const float* __restrict__ g1v, const float* __restrict__ b1v,
                    const _Float16* __restrict__ tiles, int tbase,
                    float* __restrict__ P0q, float* __restrict__ P0k, float* __restrict__ P0v,
                    float* __restrict__ P1q, float* __restrict__ P1k, float* __restrict__ P1v) {
  __shared__ _Float16 sA0[512];
  __shared__ _Float16 sA1[3][512];
  int lane = threadIdx.x;
  int r = lane & 15, hf = lane >> 4;
  int nb = blockIdx.x * 16;
  int node = nb + r;

  // deg0 norm
  {
    float f[16], nr[16];
#pragma unroll
    for (int j = 0; j < 16; j++) {
      f[j] = F0[node * 32 + hf * 16 + j];
      nr[j] = sqrtf(f[j] * f[j] + EPS);
    }
    float mu, rstd;
    norm_stats(nr, mu, rstd);
#pragma unroll
    for (int j = 0; j < 16; j++) {
      int ch = hf * 16 + j;
      float phi = eluf((nr[j] - mu) * rstd * g0v[ch] + b0v[ch]);
      sA0[r * 32 + ch] = (_Float16)(f[j] * phi / nr[j]);
    }
  }
  // deg1 norm
  {
    float fv[16][3], nr[16];
#pragma unroll
    for (int j = 0; j < 16; j++) {
      int ch = hf * 16 + j;
      float a = F1[node * 96 + ch * 3 + 0];
      float b = F1[node * 96 + ch * 3 + 1];
      float c = F1[node * 96 + ch * 3 + 2];
      fv[j][0] = a; fv[j][1] = b; fv[j][2] = c;
      nr[j] = sqrtf(a * a + b * b + c * c + EPS);
    }
    float mu, rstd;
    norm_stats(nr, mu, rstd);
#pragma unroll
    for (int j = 0; j < 16; j++) {
      int ch = hf * 16 + j;
      float phi = eluf((nr[j] - mu) * rstd * g1v[ch] + b1v[ch]);
      float sc = phi / nr[j];
      sA1[0][r * 32 + ch] = (_Float16)(fv[j][0] * sc);
      sA1[1][r * 32 + ch] = (_Float16)(fv[j][1] * sc);
      sA1[2][r * 32 + ch] = (_Float16)(fv[j][2] * sc);
    }
  }
  __syncthreads();

  int n = lane & 15, mb = (lane >> 4) * 8;
  {
    v16h a = load_a_lds(sA0, lane);
    v8f acc = wmma16(a, load_b_tile(tiles, tbase + 0, lane), zero8());
#pragma unroll
    for (int q = 0; q < 8; q++) P0q[(nb + mb + q) * 16 + n] = acc[q];
    acc = wmma16(a, load_b_tile(tiles, tbase + 1, lane), zero8());
#pragma unroll
    for (int q = 0; q < 8; q++) P0k[(nb + mb + q) * 16 + n] = acc[q];
    acc = wmma16(a, load_b_tile(tiles, tbase + 2, lane), zero8());
#pragma unroll
    for (int q = 0; q < 8; q++) P0v[(nb + mb + q) * 16 + n] = acc[q];
  }
#pragma unroll
  for (int m = 0; m < 3; m++) {
    v16h a = load_a_lds(sA1[m], lane);
    v8f acc = wmma16(a, load_b_tile(tiles, tbase + 3, lane), zero8());
#pragma unroll
    for (int q = 0; q < 8; q++) P1q[(nb + mb + q) * 48 + n * 3 + m] = acc[q];
    acc = wmma16(a, load_b_tile(tiles, tbase + 4, lane), zero8());
#pragma unroll
    for (int q = 0; q < 8; q++) P1k[(nb + mb + q) * 48 + n * 3 + m] = acc[q];
    acc = wmma16(a, load_b_tile(tiles, tbase + 5, lane), zero8());
#pragma unroll
    for (int q = 0; q < 8; q++) P1v[(nb + mb + q) * 48 + n * 3 + m] = acc[q];
  }
}

// ---------------- fused radial-net (WMMA) + CG mixing + logits + atomic max ----------------
__global__ void __launch_bounds__(128)
edge_fused_kernel(const int* __restrict__ src, const int* __restrict__ dst,
                  const float* __restrict__ rel_pos, const float* __restrict__ edge0,
                  const float* __restrict__ R1, const float* __restrict__ R1b,
                  const float* __restrict__ R2b,
                  const _Float16* __restrict__ tiles, int tr2,
                  const float* __restrict__ P0q, const float* __restrict__ P0k, const float* __restrict__ P0v,
                  const float* __restrict__ P1q, const float* __restrict__ P1k, const float* __restrict__ P1v,
                  float* __restrict__ V0e, float* __restrict__ V1e,
                  float* __restrict__ LOG, unsigned* __restrict__ Mkey) {
  __shared__ _Float16 sH[4][512];
  __shared__ float sW[4][16 * 128];
  int lane = threadIdx.x & 31;
  int w = threadIdx.x >> 5;
  int eb = (blockIdx.x * 4 + w) * 16;
  int r = lane & 15, hf = lane >> 4;

  // phase A: radial hidden (16 edges x 32 hidden) -> f16 A tile
  {
    int e = eb + r;
    float x = rel_pos[e * 3 + 0], y = rel_pos[e * 3 + 1], z = rel_pos[e * 3 + 2];
    float d = sqrtf(x * x + y * y + z * z + EPS);
    float e0 = edge0[e * 2 + 0], e1 = edge0[e * 2 + 1];
#pragma unroll
    for (int j = 0; j < 16; j++) {
      int c = hf * 16 + j;
      float h = R1[c * 3 + 0] * d + R1[c * 3 + 1] * e0 + R1[c * 3 + 2] * e1 + R1b[c];
      sH[w][r * 32 + c] = (_Float16)fmaxf(h, 0.f);
    }
  }
  __syncthreads();
  // 8 WMMAs: hidden(16x32) @ R2^T(32x128) -> w tile in LDS
  {
    v16h a = load_a_lds(sH[w], lane);
    int n = lane & 15, mb = (lane >> 4) * 8;
#pragma unroll
    for (int t = 0; t < 8; t++) {
      v8f acc = zero8();
      float bias = R2b[t * 16 + n];
#pragma unroll
      for (int q = 0; q < 8; q++) acc[q] = bias;
      acc = wmma16(a, load_b_tile(tiles, tr2 + t, lane), acc);
#pragma unroll
      for (int q = 0; q < 8; q++) sW[w][(mb + q) * 128 + t * 16 + n] = acc[q];
    }
  }
  __syncthreads();
  // phase B: 2 lanes per edge, 8 channels each
  {
    int er = lane & 15;
    int e = eb + er;
    int cp = (lane >> 4) * 8;
    int s = src[e], dd = dst[e];
    float x = rel_pos[e * 3 + 0], y = rel_pos[e * 3 + 1], z = rel_pos[e * 3 + 2];
    float d = sqrtf(x * x + y * y + z * z + EPS);
    float Y[3] = { x / d, y / d, z / d };
    const float* wr = &sW[w][er * 128];
    float lacc[4] = { 0.f, 0.f, 0.f, 0.f };
#pragma unroll
    for (int j = 0; j < 8; j++) {
      int c = cp + j;
      float q0 = P0q[dd * 16 + c];
      float a0 = P0k[s * 16 + c];
      float b0 = P0v[s * 16 + c];
      float q1[3], a1[3], b1[3];
#pragma unroll
      for (int m = 0; m < 3; m++) {
        q1[m] = P1q[dd * 48 + c * 3 + m];
        a1[m] = P1k[s * 48 + c * 3 + m];
        b1[m] = P1v[s * 48 + c * 3 + m];
      }
      float w0 = wr[0 * 16 + c], w1 = wr[1 * 16 + c], w2 = wr[2 * 16 + c], w3 = wr[3 * 16 + c];
      float w4 = wr[4 * 16 + c], w5 = wr[5 * 16 + c], w6 = wr[6 * 16 + c], w7 = wr[7 * 16 + c];
      float a1Y = a1[0] * Y[0] + a1[1] * Y[1] + a1[2] * Y[2];
      float b1Y = b1[0] * Y[0] + b1[1] * Y[1] + b1[2] * Y[2];
      float k0 = w0 * a0 + w1 * a1Y;
      float v0 = w4 * b0 + w5 * b1Y;
      float lg = q0 * k0;
      V0e[(size_t)e * 16 + c] = v0;
#pragma unroll
      for (int m = 0; m < 3; m++) {
        float k1 = w2 * a1[m] + w3 * a0 * Y[m];
        float v1 = w6 * b1[m] + w7 * b0 * Y[m];
        lg += q1[m] * k1;
        V1e[(size_t)e * 48 + c * 3 + m] = v1;
      }
      lacc[j >> 1] += lg;
    }
    const float scale = 0.35355339059327373f; // 1/sqrt(DH*4)
#pragma unroll
    for (int hh = 0; hh < 4; hh++) {
      int head = (cp >> 1) + hh;
      float lv = lacc[hh] * scale;
      LOG[(size_t)e * 8 + head] = lv;
      atomicMax(&Mkey[dd * 8 + head], fkey(lv));
    }
  }
}

__global__ void softmax_pass2_kernel(const int* __restrict__ dst, const unsigned* __restrict__ Mkey,
                                     float* __restrict__ LOG, float* __restrict__ Z) {
  int i = blockIdx.x * blockDim.x + threadIdx.x;
  if (i >= NE * 8) return;
  int e = i >> 3, h = i & 7;
  int dd = dst[e];
  float m = fkey_dec(Mkey[dd * 8 + h]);
  float ex = expf(LOG[i] - m);
  LOG[i] = ex;
  atomicAdd(&Z[dd * 8 + h], ex);
}

__global__ void scatter_pass3_kernel(const int* __restrict__ dst, const float* __restrict__ LOG,
                                     const float* __restrict__ Z,
                                     const float* __restrict__ V0e, const float* __restrict__ V1e,
                                     float* __restrict__ O0, float* __restrict__ O1) {
  int i = blockIdx.x * blockDim.x + threadIdx.x;
  if (i >= NE * 2) return;
  int e = i >> 1, cp = (i & 1) * 8;
  int dd = dst[e];
#pragma unroll
  for (int j = 0; j < 8; j++) {
    int c = cp + j;
    int h = c >> 1;
    float attn = LOG[(size_t)e * 8 + h] / (Z[dd * 8 + h] + 1e-9f);
    atomicAdd(&O0[dd * 16 + c], attn * V0e[(size_t)e * 16 + c]);
#pragma unroll
    for (int m = 0; m < 3; m++)
      atomicAdd(&O1[dd * 48 + c * 3 + m], attn * V1e[(size_t)e * 48 + c * 3 + m]);
  }
}

// ---------------- aggregation: f = o@Wo^T + f@Ws^T (chained WMMA accumulate) ----------------
__global__ void __launch_bounds__(32)
aggregate_kernel(const float* __restrict__ O0, const float* __restrict__ O1,
                 const _Float16* __restrict__ tiles,
                 int two0, int tws0, int two1, int tws1,
                 float* __restrict__ F0, float* __restrict__ F1) {
  __shared__ _Float16 sA[512];
  int lane = threadIdx.x;
  int r = lane & 15, hf = lane >> 4;
  int nb = blockIdx.x * 16;
  int node = nb + r;
  int n = r, mb = hf * 8;

  // deg0
  v8f acc0[2];
#pragma unroll
  for (int j = 0; j < 16; j++) {
    int ch = hf * 16 + j;
    sA[r * 32 + ch] = (_Float16)((ch < 16) ? O0[node * 16 + ch] : 0.f);
  }
  __syncthreads();
  {
    v16h a = load_a_lds(sA, lane);
    for (int t = 0; t < 2; t++) acc0[t] = wmma16(a, load_b_tile(tiles, two0 + t, lane), zero8());
  }
  __syncthreads();
#pragma unroll
  for (int j = 0; j < 16; j++) { int ch = hf * 16 + j; sA[r * 32 + ch] = (_Float16)F0[node * 32 + ch]; }
  __syncthreads();
  {
    v16h a = load_a_lds(sA, lane);
    for (int t = 0; t < 2; t++) acc0[t] = wmma16(a, load_b_tile(tiles, tws0 + t, lane), acc0[t]);
  }
  for (int t = 0; t < 2; t++)
#pragma unroll
    for (int q = 0; q < 8; q++) F0[(nb + mb + q) * 32 + t * 16 + n] = acc0[t][q];

  // deg1 (per component)
  for (int m = 0; m < 3; m++) {
    __syncthreads();
#pragma unroll
    for (int j = 0; j < 16; j++) {
      int ch = hf * 16 + j;
      sA[r * 32 + ch] = (_Float16)((ch < 16) ? O1[node * 48 + ch * 3 + m] : 0.f);
    }
    __syncthreads();
    v8f acc[2];
    {
      v16h a = load_a_lds(sA, lane);
      for (int t = 0; t < 2; t++) acc[t] = wmma16(a, load_b_tile(tiles, two1 + t, lane), zero8());
    }
    __syncthreads();
#pragma unroll
    for (int j = 0; j < 16; j++) { int ch = hf * 16 + j; sA[r * 32 + ch] = (_Float16)F1[node * 96 + ch * 3 + m]; }
    __syncthreads();
    {
      v16h a = load_a_lds(sA, lane);
      for (int t = 0; t < 2; t++) acc[t] = wmma16(a, load_b_tile(tiles, tws1 + t, lane), acc[t]);
    }
    for (int t = 0; t < 2; t++)
#pragma unroll
      for (int q = 0; q < 8; q++) F1[(nb + mb + q) * 96 + (t * 16 + n) * 3 + m] = acc[t][q];
  }
}

// ---------------- final projection: g = f@Wf^T + out0 ----------------
__global__ void __launch_bounds__(32)
final_proj_kernel(const float* __restrict__ F0, const float* __restrict__ F1,
                  const float* __restrict__ out0_0, const float* __restrict__ out0_1,
                  const _Float16* __restrict__ tiles, int twf0, int twf1,
                  float* __restrict__ g0, float* __restrict__ g1) {
  __shared__ _Float16 sA[512];
  int lane = threadIdx.x;
  int r = lane & 15, hf = lane >> 4;
  int nb = blockIdx.x * 16;
  int node = nb + r;
  int n = r, mb = hf * 8;

#pragma unroll
  for (int j = 0; j < 16; j++) { int ch = hf * 16 + j; sA[r * 32 + ch] = (_Float16)F0[node * 32 + ch]; }
  __syncthreads();
  {
    v16h a = load_a_lds(sA, lane);
    for (int t = 0; t < 2; t++) {
      v8f acc = wmma16(a, load_b_tile(tiles, twf0 + t, lane), zero8());
#pragma unroll
      for (int q = 0; q < 8; q++) {
        int idx = (nb + mb + q) * 32 + t * 16 + n;
        g0[idx] = acc[q] + out0_0[idx];
      }
    }
  }
  for (int m = 0; m < 3; m++) {
    __syncthreads();
#pragma unroll
    for (int j = 0; j < 16; j++) { int ch = hf * 16 + j; sA[r * 32 + ch] = (_Float16)F1[node * 96 + ch * 3 + m]; }
    __syncthreads();
    v16h a = load_a_lds(sA, lane);
    for (int t = 0; t < 2; t++) {
      v8f acc = wmma16(a, load_b_tile(tiles, twf1 + t, lane), zero8());
#pragma unroll
      for (int q = 0; q < 8; q++) {
        int idx = (nb + mb + q) * 96 + (t * 16 + n) * 3 + m;
        g1[idx] = acc[q] + out0_1[idx];
      }
    }
  }
}

// ---------------- linear_module (h1 branch is dead code in reference) ----------------
__global__ void __launch_bounds__(32)
linear_module_kernel(const float* __restrict__ g0,
                     const float* __restrict__ lm_g0, const float* __restrict__ lm_b0,
                     const _Float16* __restrict__ tiles, int tlm0, int tlmf,
                     float* __restrict__ sc_out, float* __restrict__ sc0_out) {
  __shared__ _Float16 sA[512];
  __shared__ float sF[512];
  int lane = threadIdx.x;
  int r = lane & 15, hf = lane >> 4;
  int nb = blockIdx.x * 16;
  int node = nb + r;
  int n = r, mb = hf * 8;

  float h[16];
#pragma unroll
  for (int j = 0; j < 16; j++) h[j] = g0[node * 32 + hf * 16 + j];

  for (int it = 0; it < 3; it++) {
    float nr[16];
#pragma unroll
    for (int j = 0; j < 16; j++) nr[j] = sqrtf(h[j] * h[j] + EPS);
    float mu, rstd;
    norm_stats(nr, mu, rstd);
    __syncthreads();
#pragma unroll
    for (int j = 0; j < 16; j++) {
      int ch = hf * 16 + j;
      float phi = eluf((nr[j] - mu) * rstd * lm_g0[it * 32 + ch] + lm_b0[it * 32 + ch]);
      sA[r * 32 + ch] = (_Float16)(h[j] * phi / nr[j]);
    }
    __syncthreads();
    v8f acc[2];
    {
      v16h a = load_a_lds(sA, lane);
      for (int t = 0; t < 2; t++) acc[t] = wmma16(a, load_b_tile(tiles, tlm0 + it * 2 + t, lane), zero8());
    }
    __syncthreads();
    for (int t = 0; t < 2; t++)
#pragma unroll
      for (int q = 0; q < 8; q++) sF[(mb + q) * 32 + t * 16 + n] = acc[t][q];
    __syncthreads();
#pragma unroll
    for (int j = 0; j < 16; j++) h[j] = sF[r * 32 + hf * 16 + j];
  }
  // final norm + projection to 14 outputs
  {
    float nr[16];
#pragma unroll
    for (int j = 0; j < 16; j++) nr[j] = sqrtf(h[j] * h[j] + EPS);
    float mu, rstd;
    norm_stats(nr, mu, rstd);
    __syncthreads();
#pragma unroll
    for (int j = 0; j < 16; j++) {
      int ch = hf * 16 + j;
      float phi = eluf((nr[j] - mu) * rstd * lm_g0[3 * 32 + ch] + lm_b0[3 * 32 + ch]);
      sA[r * 32 + ch] = (_Float16)(h[j] * phi / nr[j]);
    }
    __syncthreads();
    v16h a = load_a_lds(sA, lane);
    v8f acc = wmma16(a, load_b_tile(tiles, tlmf, lane), zero8());
#pragma unroll
    for (int q = 0; q < 8; q++) {
      float x = acc[q];
      float y = __shfl_xor(x, 1);   // pair partner column (n^1) lives in lane^1
      if (n < 14) {
        int nd = nb + mb + q;
        sc0_out[nd * 14 + n] = x;
        sc_out[nd * 14 + n] = x * rsqrtf(x * x + y * y + EPS);
      }
    }
  }
}

extern "C" void kernel_launch(void* const* d_in, const int* in_sizes, int n_in,
                              void* d_out, int out_size, void* d_ws, size_t ws_size,
                              hipStream_t stream) {
  (void)in_sizes; (void)n_in; (void)out_size; (void)ws_size;
  const int* src = (const int*)d_in[0];
  const int* dst = (const int*)d_in[1];
  const float* rel_pos = (const float*)d_in[2];
  const float* node0 = (const float*)d_in[3];
  const float* node1 = (const float*)d_in[4];
  const float* edge0 = (const float*)d_in[5];
  const float* out0_0 = (const float*)d_in[6];
  const float* out0_1 = (const float*)d_in[7];
  const float* ln_g0 = (const float*)d_in[8];
  const float* ln_b0 = (const float*)d_in[9];
  const float* ln_g1 = (const float*)d_in[10];
  const float* ln_b1 = (const float*)d_in[11];
  const float* Wq0 = (const float*)d_in[12];
  const float* Wq1 = (const float*)d_in[13];
  const float* Wk0 = (const float*)d_in[14];
  const float* Wk1 = (const float*)d_in[15];
  const float* Wv0 = (const float*)d_in[16];
  const float* Wv1 = (const float*)d_in[17];
  const float* R1 = (const float*)d_in[18];
  const float* R1b = (const float*)d_in[19];
  const float* R2 = (const float*)d_in[20];
  const float* R2b = (const float*)d_in[21];
  const float* Wo0 = (const float*)d_in[22];
  const float* Wo1 = (const float*)d_in[23];
  const float* Ws0 = (const float*)d_in[24];
  const float* Ws1 = (const float*)d_in[25];
  const float* Wf0 = (const float*)d_in[26];
  const float* Wf1 = (const float*)d_in[27];
  const float* lm_g0 = (const float*)d_in[28];
  const float* lm_b0 = (const float*)d_in[29];
  const float* lm_W0 = (const float*)d_in[32];
  const float* lm_Wf0 = (const float*)d_in[34];

  char* ws = (char*)d_ws;
  size_t off = 0;
  auto alloc = [&](size_t bytes) { char* p = ws + off; off += (bytes + 255) & ~(size_t)255; return p; };

  _Float16* tiles = (_Float16*)alloc(55 * 1024);
  float* F0 = (float*)alloc((size_t)NN * 32 * 4);
  float* F1 = (float*)alloc((size_t)NN * 96 * 4);
  float* P0q = (float*)alloc((size_t)NN * 16 * 4);
  float* P0k = (float*)alloc((size_t)NN * 16 * 4);
  float* P0v = (float*)alloc((size_t)NN * 16 * 4);
  float* P1q = (float*)alloc((size_t)NN * 48 * 4);
  float* P1k = (float*)alloc((size_t)NN * 48 * 4);
  float* P1v = (float*)alloc((size_t)NN * 48 * 4);
  unsigned* Mkey = (unsigned*)alloc((size_t)NN * 8 * 4);   // contiguous scatter region start
  float* Z = (float*)alloc((size_t)NN * 8 * 4);
  float* O0 = (float*)alloc((size_t)NN * 16 * 4);
  float* O1 = (float*)alloc((size_t)NN * 48 * 4);
  float* V0e = (float*)alloc((size_t)NE * 16 * 4);
  float* V1e = (float*)alloc((size_t)NE * 48 * 4);
  float* LOG = (float*)alloc((size_t)NE * 8 * 4);

  auto pack = [&](const float* W, int rows, int cols, int n0, int k0, int tile) {
    pack_b_kernel<<<1, 32, 0, stream>>>(W, rows, cols, n0, k0, tiles + (size_t)tile * 512);
  };
  for (int l = 0; l < 2; l++) {
    int base = l * 22;
    pack(Wq0 + l * 16 * 32, 16, 32, 0, 0, base + 0);
    pack(Wk0 + l * 16 * 32, 16, 32, 0, 0, base + 1);
    pack(Wv0 + l * 16 * 32, 16, 32, 0, 0, base + 2);
    pack(Wq1 + l * 16 * 32, 16, 32, 0, 0, base + 3);
    pack(Wk1 + l * 16 * 32, 16, 32, 0, 0, base + 4);
    pack(Wv1 + l * 16 * 32, 16, 32, 0, 0, base + 5);
    for (int t = 0; t < 8; t++) pack(R2 + l * 128 * 32, 128, 32, t * 16, 0, base + 6 + t);
    for (int t = 0; t < 2; t++) pack(Wo0 + l * 32 * 16, 32, 16, t * 16, 0, base + 14 + t);
    for (int t = 0; t < 2; t++) pack(Ws0 + l * 32 * 32, 32, 32, t * 16, 0, base + 16 + t);
    for (int t = 0; t < 2; t++) pack(Wo1 + l * 32 * 16, 32, 16, t * 16, 0, base + 18 + t);
    for (int t = 0; t < 2; t++) pack(Ws1 + l * 32 * 32, 32, 32, t * 16, 0, base + 20 + t);
  }
  for (int t = 0; t < 2; t++) pack(Wf0, 32, 32, t * 16, 0, 44 + t);
  for (int t = 0; t < 2; t++) pack(Wf1, 32, 32, t * 16, 0, 46 + t);
  for (int i = 0; i < 3; i++)
    for (int t = 0; t < 2; t++) pack(lm_W0 + i * 32 * 32, 32, 32, t * 16, 0, 48 + i * 2 + t);
  pack(lm_Wf0, 14, 32, 0, 0, 54);

  init_f_kernel<<<(NN * 128 + 255) / 256, 256, 0, stream>>>(node0, node1, F0, F1);

  for (int l = 0; l < 2; l++) {
    int base = l * 22;
    node_project_kernel<<<NN / 16, 32, 0, stream>>>(
        F0, F1, ln_g0 + l * 32, ln_b0 + l * 32, ln_g1 + l * 32, ln_b1 + l * 32,
        tiles, base, P0q, P0k, P0v, P1q, P1k, P1v);
    int zn = NN * (8 + 8 + 16 + 48);
    zero_kernel<<<(zn + 255) / 256, 256, 0, stream>>>((unsigned*)Mkey, zn);
    edge_fused_kernel<<<NE / 64, 128, 0, stream>>>(
        src, dst, rel_pos, edge0, R1 + l * 96, R1b + l * 32, R2b + l * 128,
        tiles, base + 6, P0q, P0k, P0v, P1q, P1k, P1v, V0e, V1e, LOG, Mkey);
    softmax_pass2_kernel<<<(NE * 8) / 256, 256, 0, stream>>>(dst, Mkey, LOG, Z);
    scatter_pass3_kernel<<<(NE * 2) / 256, 256, 0, stream>>>(dst, LOG, Z, V0e, V1e, O0, O1);
    aggregate_kernel<<<NN / 16, 32, 0, stream>>>(
        O0, O1, tiles, base + 14, base + 16, base + 18, base + 20, F0, F1);
  }

  float* g0 = (float*)d_out;
  float* g1 = g0 + (size_t)NN * 32;
  float* sc = g0 + (size_t)NN * (32 + 96);
  float* sc0 = sc + (size_t)NN * 14;
  final_proj_kernel<<<NN / 16, 32, 0, stream>>>(F0, F1, out0_0, out0_1, tiles, 44, 46, g0, g1);
  linear_module_kernel<<<NN / 16, 32, 0, stream>>>(g0, lm_g0, lm_b0, tiles, 48, 54, sc, sc0);
}
